// RealBounce_5016521801860
// MI455X (gfx1250) — matrix operand order repaired and checked by reading, not compiled, not run on hardware
//
#include <hip/hip_runtime.h>
#include <hip/hip_bf16.h>
#include <math.h>

typedef __attribute__((ext_vector_type(16))) _Float16 v16h;
typedef __attribute__((ext_vector_type(8)))  _Float16 v8h;
typedef __attribute__((ext_vector_type(8)))  float    v8f;

// ---------------- WMMA fragment helpers (16x16x32 f16 -> f32) ----------------
// A (16x32 MxK) fragment for lane (m = lane&15, hs = lane>>4) is two contiguous
// 16-byte chunks of the row-major activation row:
//   K = [kbase + 8*hs .. +7]  and  K = [kbase + 16 + 8*hs .. +7]
__device__ __forceinline__ v16h frag_a_lds(const _Float16* base, int stride, int kbase) {
  const int lane = threadIdx.x & 31;
  const int mm = lane & 15, hs = lane >> 4;
  const _Float16* p = base + mm * stride + kbase + (hs << 3);
  v8h lo = *(const v8h*)p;
  v8h hi = *(const v8h*)(p + 16);
  return __builtin_shufflevector(lo, hi, 0, 1, 2, 3, 4, 5, 6, 7,
                                 8, 9, 10, 11, 12, 13, 14, 15);
}

// B (32x16 KxN) fragment from TRANSPOSED weights wT[n][Kpad]:
//   K = [kbase + 8*hs .. +7]  and  K = [kbase + 16 + 8*hs .. +7] (matches A).
__device__ __forceinline__ v16h frag_b_ldsT(const _Float16* wT, int Kpad, int kbase, int n) {
  const int lane = threadIdx.x & 31;
  const int hs = lane >> 4;
  const _Float16* p = wT + n * Kpad + kbase + (hs << 3);
  v8h lo = *(const v8h*)p;
  v8h hi = *(const v8h*)(p + 16);
  return __builtin_shufflevector(lo, hi, 0, 1, 2, 3, 4, 5, 6, 7,
                                 8, 9, 10, 11, 12, 13, 14, 15);
}

__device__ __forceinline__ v8f wmma16(v16h a, v16h b, v8f c) {
  return __builtin_amdgcn_wmma_f32_16x16x32_f16(false, a, false, b, (short)0, c, false, false);
}

__device__ __forceinline__ float sigmoidf_(float x) { return 1.0f / (1.0f + expf(-x)); }

// ========================= Kernel 1: diffuse MLP over N points =========================
#define K1_WAVES 4
__global__ __launch_bounds__(128) void diffuse_mlp_kernel(
    const float* __restrict__ xyzs, const float* __restrict__ viewdirs,
    const float* __restrict__ app,
    const float* __restrict__ Wd1, const float* __restrict__ bd1,
    const float* __restrict__ Wd2, const float* __restrict__ bd2,
    float* __restrict__ out, float* __restrict__ r1_ws)
{
  __shared__ _Float16 sWd1T[64 * 96];                // 12 KB, transposed, K padded 70->96
  __shared__ _Float16 sWd2T[16 * 64];                //  2 KB, transposed, N padded 8->16
  __shared__ _Float16 inbuf[K1_WAVES][16 * 96];      // 12 KB
  __shared__ _Float16 hbuf[K1_WAVES][16 * 64];       //  8 KB

  const int tid = threadIdx.x;
  const int wv = tid >> 5;
  const int lane = tid & 31;
  const int ncol = lane & 15, hs = lane >> 4;

  // ---- preload per-lane biases into registers (avoid vmem stalls in epilogues) ----
  float bias1[4];
#pragma unroll
  for (int ctl = 0; ctl < 4; ++ctl) bias1[ctl] = bd1[ctl * 16 + ncol];
  const float bias2 = (ncol < 8) ? bd2[ncol] : 0.0f;

  // stage Wd1^T (70x64 -> [64][96])
  for (int i = tid; i < 64 * 96; i += 128) {
    const int n = i / 96, k = i - n * 96;
    sWd1T[i] = (k < 70) ? (_Float16)Wd1[k * 64 + n] : (_Float16)0.0f;
  }
  // stage Wd2^T (64x8 -> [16][64])
  for (int i = tid; i < 16 * 64; i += 128) {
    const int n = i >> 6, k = i & 63;
    sWd2T[i] = (n < 8) ? (_Float16)Wd2[k * 8 + n] : (_Float16)0.0f;
  }

  const int rowbase = (blockIdx.x * K1_WAVES + wv) * 16;

  if (lane < 16) {
    const int row = rowbase + lane;
    _Float16* dst = &inbuf[wv][lane * 96];
    float vx = viewdirs[row * 3 + 0], vy = viewdirs[row * 3 + 1], vz = viewdirs[row * 3 + 2];
    const float inv = 1.0f / (sqrtf(vx * vx + vy * vy + vz * vz) + 1e-8f);
    dst[0] = (_Float16)xyzs[row * 3 + 0];
    dst[1] = (_Float16)xyzs[row * 3 + 1];
    dst[2] = (_Float16)xyzs[row * 3 + 2];
    dst[3] = (_Float16)(vx * inv); dst[4] = (_Float16)(vy * inv); dst[5] = (_Float16)(vz * inv);
    const float4* arow = (const float4*)(app + row * 64);
#pragma unroll
    for (int j = 0; j < 16; ++j) {
      const float4 a4 = arow[j];
      dst[6 + 4 * j + 0] = (_Float16)a4.x; dst[6 + 4 * j + 1] = (_Float16)a4.y;
      dst[6 + 4 * j + 2] = (_Float16)a4.z; dst[6 + 4 * j + 3] = (_Float16)a4.w;
    }
#pragma unroll
    for (int j = 70; j < 96; ++j) dst[j] = (_Float16)0.0f;
  }
  __syncthreads();

  // layer 1: h = relu(in @ Wd1 + bd1), 64 outputs = 4 col tiles, K=96
  for (int ctl = 0; ctl < 4; ++ctl) {
    v8f c = {};
#pragma unroll
    for (int kc = 0; kc < 3; ++kc) {
      v16h aa = frag_a_lds(inbuf[wv], 96, kc * 32);
      v16h bb = frag_b_ldsT(sWd1T, 96, kc * 32, ctl * 16 + ncol);
      c = wmma16(aa, bb, c);
    }
    const int n = ctl * 16 + ncol;
    const float bias = bias1[ctl];
#pragma unroll
    for (int r = 0; r < 8; ++r) {
      const float h = c[r] + bias;
      hbuf[wv][(hs * 8 + r) * 64 + n] = (_Float16)(h > 0.0f ? h : 0.0f);
    }
  }
  __syncthreads();

  // layer 2: o = h @ Wd2 + bd2 (8 outputs, one col tile), K=64
  {
    v8f c = {};
#pragma unroll
    for (int kc = 0; kc < 2; ++kc) {
      v16h aa = frag_a_lds(hbuf[wv], 64, kc * 32);
      v16h bb = frag_b_ldsT(sWd2T, 64, kc * 32, ncol);
      c = wmma16(aa, bb, c);
    }
    if (ncol < 3) {
#pragma unroll
      for (int r = 0; r < 8; ++r) {
        const int row = rowbase + hs * 8 + r;
        out[row * 3 + ncol] = sigmoidf_(c[r] + bias2);
      }
    } else if (ncol == 6) {
#pragma unroll
      for (int r = 0; r < 8; ++r) {
        const int row = rowbase + hs * 8 + r;
        r1_ws[row] = sigmoidf_(c[r] + bias2) * 0.98f + 0.01f;
      }
    }
  }
}

// ========================= Kernel 2: bounce shading (1 wave = 1 bounce point) =========================
#define K2_WAVES 4
__global__ __launch_bounds__(128) void bounce_kernel(
    const float* __restrict__ app, const float* __restrict__ viewdirs,
    const float* __restrict__ normals, const float* __restrict__ noise,
    const float* __restrict__ u,
    const float* __restrict__ Wb1, const float* __restrict__ bb1,
    const float* __restrict__ Wb2, const float* __restrict__ bb2,
    const float* __restrict__ We1, const float* __restrict__ be1,
    const float* __restrict__ We2, const float* __restrict__ be2,
    const int* __restrict__ bounce_idx, const unsigned char* __restrict__ ray_mask,
    const float* __restrict__ r1_ws, float* __restrict__ out)
{
  __shared__ _Float16 sWb1T[128 * 96];               // 24 KB, transposed, K padded 80->96
  __shared__ _Float16 sWb2T[16 * 128];               //  4 KB, transposed, N padded 3->16
  __shared__ _Float16 sWe1T[64 * 32];                //  4 KB, transposed, K padded 4->32
  __shared__ _Float16 sWe2T[16 * 64];                //  2 KB, transposed, N padded 3->16
  __shared__ _Float16 featbuf[K2_WAVES][16 * 96];    // 12 KB (ef stride 32 / bin stride 96)
  __shared__ _Float16 hbuf[K2_WAVES][16 * 128];      // 16 KB (stride 64 / 128)
  __shared__ _Float16 sIncoming[K2_WAVES][32][3];    // 768 B
  __shared__ _Float16 snaf[K2_WAVES][64];            // 512 B
  // total: 64,768 B

  const int tid = threadIdx.x, wv = tid >> 5, lane = tid & 31;
  const int nb = blockIdx.x * K2_WAVES + wv;
  const int ncol = lane & 15, hs = lane >> 4;

  // ---- preload per-lane biases into registers ----
  float be1v[4], bb1v[8];
#pragma unroll
  for (int ctl = 0; ctl < 4; ++ctl) be1v[ctl] = be1[ctl * 16 + ncol];
#pragma unroll
  for (int ctl = 0; ctl < 8; ++ctl) bb1v[ctl] = bb1[ctl * 16 + ncol];
  const float be2v = (ncol < 3) ? be2[ncol] : 0.0f;
  const float bb2v = (ncol < 3) ? bb2[ncol] : 0.0f;

  // ---- stage transposed weights ----
  for (int i = tid; i < 128 * 96; i += 128) {
    const int n = i / 96, k = i - n * 96;
    sWb1T[i] = (k < 80) ? (_Float16)Wb1[k * 128 + n] : (_Float16)0.0f;
  }
  for (int i = tid; i < 16 * 128; i += 128) {
    const int n = i >> 7, k = i & 127;
    sWb2T[i] = (n < 3) ? (_Float16)Wb2[k * 3 + n] : (_Float16)0.0f;
  }
  for (int i = tid; i < 64 * 32; i += 128) {
    const int n = i >> 5, k = i & 31;
    sWe1T[i] = (k < 4) ? (_Float16)We1[k * 64 + n] : (_Float16)0.0f;
  }
  for (int i = tid; i < 16 * 64; i += 128) {
    const int n = i >> 6, k = i & 63;
    sWe2T[i] = (n < 3) ? (_Float16)We2[k * 3 + n] : (_Float16)0.0f;
  }

  const int idx = bounce_idx[nb];

  // stage noised app features (shared by all 32 rays of this bounce point)
  snaf[wv][lane]      = (_Float16)(app[idx * 64 + lane]      + 0.1f * noise[idx * 64 + lane]);
  snaf[wv][lane + 32] = (_Float16)(app[idx * 64 + lane + 32] + 0.1f * noise[idx * 64 + lane + 32]);

  // --- per-bounce-point quantities (redundant in every lane) ---
  float nx = normals[idx * 3 + 0], ny = normals[idx * 3 + 1], nz = normals[idx * 3 + 2];
  { const float inv = 1.f / (sqrtf(nx * nx + ny * ny + nz * nz) + 1e-8f); nx *= inv; ny *= inv; nz *= inv; }
  float vx = viewdirs[idx * 3 + 0], vy = viewdirs[idx * 3 + 1], vz = viewdirs[idx * 3 + 2];
  { const float inv = 1.f / (sqrtf(vx * vx + vy * vy + vz * vz) + 1e-8f); vx *= inv; vy *= inv; vz *= inv; }
  const float bvx = -vx, bvy = -vy, bvz = -vz;
  const float br1 = r1_ws[idx];
  const float alpha = br1 * br1;

  float ux, uy, uz;
  if (fabsf(nz) < 0.99f) { ux = 0.f; uy = 0.f; uz = 1.f; } else { ux = 1.f; uy = 0.f; uz = 0.f; }
  float tx = uy * nz - uz * ny, ty = uz * nx - ux * nz, tz = ux * ny - uy * nx;
  { const float inv = 1.f / (sqrtf(tx * tx + ty * ty + tz * tz) + 1e-8f); tx *= inv; ty *= inv; tz *= inv; }
  const float btx = ny * tz - nz * ty, bty = nz * tx - nx * tz, btz = nx * ty - ny * tx;

  // --- per-ray GGX math: lane m owns ray m ---
  const int m = lane;
  const float2 u2v = ((const float2*)u)[nb * 32 + m];
  const float u1 = u2v.x, u2 = u2v.y;
  const float a = alpha;
  float ct2 = (1.f - u1) / (1.f + (a * a - 1.f) * u1);
  ct2 = fminf(fmaxf(ct2, 0.f), 1.f);
  const float ctv = sqrtf(ct2);
  const float st2 = fminf(fmaxf(1.f - ctv * ctv, 0.f), 1.f);
  const float stv = sqrtf(st2);
  const float phi = 6.283185307179586f * u2;
  const float hlx = stv * cosf(phi), hly = stv * sinf(phi), hlz = ctv;
  const float Hwx = tx * hlx + btx * hly + nx * hlz;
  const float Hwy = ty * hlx + bty * hly + ny * hlz;
  const float Hwz = tz * hlx + btz * hly + nz * hlz;
  const float vdh = bvx * Hwx + bvy * Hwy + bvz * Hwz;
  float Lx = 2.f * vdh * Hwx - bvx, Ly = 2.f * vdh * Hwy - bvy, Lz = 2.f * vdh * Hwz - bvz;
  { const float inv = 1.f / (sqrtf(Lx * Lx + Ly * Ly + Lz * Lz) + 1e-8f); Lx *= inv; Ly *= inv; Lz *= inv; }
  float Hhx = bvx + Lx, Hhy = bvy + Ly, Hhz = bvz + Lz;
  { const float inv = 1.f / (sqrtf(Hhx * Hhx + Hhy * Hhy + Hhz * Hhz) + 1e-8f); Hhx *= inv; Hhy *= inv; Hhz *= inv; }
  const float ea = br1, am = ea * ea, am2 = am * am;
  float ndh = fminf(fmaxf(nx * Hhx + ny * Hhy + nz * Hhz, 1e-4f), 1.f);
  const float den = ndh * ndh * (am2 - 1.f) + 1.f;
  const float D = am2 / (3.14159265358979f * den * den + 1e-8f);
  const float mip = -logf(D * ndh * 32.f + 1e-8f);
  const float dvx = tx * Lx + ty * Ly + tz * Lz;
  const float dvy = btx * Lx + bty * Ly + btz * Lz;
  const float dvz = nx * Lx + ny * Ly + nz * Lz;
  const float hvx = tx * Hhx + ty * Hhy + tz * Hhz;
  const float hvy = btx * Hhx + bty * Hhy + btz * Hhz;
  const float hvz = nx * Hhx + ny * Hhy + nz * Hhz;

  const bool maskb = ray_mask[nb * 32 + m] != 0;
  const unsigned long long bal = __ballot(maskb);     // all lanes hold full 32-ray mask
  const float invrc = 1.0f / ((float)__popcll(bal) + 1e-8f);
  __syncthreads();

  // ---- incoming MLP: softplus(relu([L, mip] @ We1 + be1) @ We2 + be2) ----
  for (int t = 0; t < 2; ++t) {
    if ((m >> 4) == t) {
      _Float16* dst = &featbuf[wv][(m & 15) * 32];
      dst[0] = (_Float16)Lx; dst[1] = (_Float16)Ly; dst[2] = (_Float16)Lz; dst[3] = (_Float16)mip;
#pragma unroll
      for (int j = 4; j < 32; ++j) dst[j] = (_Float16)0.0f;
    }
    __syncthreads();
    for (int ctl = 0; ctl < 4; ++ctl) {
      v8f c = {};
      v16h aa = frag_a_lds(featbuf[wv], 32, 0);
      v16h bb = frag_b_ldsT(sWe1T, 32, 0, ctl * 16 + ncol);
      c = wmma16(aa, bb, c);
      const int n = ctl * 16 + ncol;
      const float bias = be1v[ctl];
#pragma unroll
      for (int r = 0; r < 8; ++r) {
        const float h = c[r] + bias;
        hbuf[wv][(hs * 8 + r) * 64 + n] = (_Float16)(h > 0.f ? h : 0.f);
      }
    }
    __syncthreads();
    {
      v8f c = {};
#pragma unroll
      for (int kc = 0; kc < 2; ++kc) {
        v16h aa = frag_a_lds(hbuf[wv], 64, kc * 32);
        v16h bb = frag_b_ldsT(sWe2T, 64, kc * 32, ncol);
        c = wmma16(aa, bb, c);
      }
      if (ncol < 3) {
#pragma unroll
        for (int r = 0; r < 8; ++r) {
          const float x = c[r] + be2v;
          sIncoming[wv][t * 16 + hs * 8 + r][ncol] =
              (_Float16)((x > 20.f) ? x : logf(1.f + expf(x)));
        }
      }
    }
    __syncthreads();
  }

  // ---- bw MLP: sigmoid(relu(bin @ Wb1 + bb1) @ Wb2 + bb2), fused masked reduce ----
  float acc = 0.f;
  for (int t = 0; t < 2; ++t) {
    if ((m >> 4) == t) {
      _Float16* dst = &featbuf[wv][(m & 15) * 96];
      dst[0] = (_Float16)bvx; dst[1] = (_Float16)bvy; dst[2] = (_Float16)bvz;
      dst[3] = (_Float16)Lx;  dst[4] = (_Float16)Ly;  dst[5] = (_Float16)Lz;
      dst[6] = (_Float16)nx;  dst[7] = (_Float16)ny;  dst[8] = (_Float16)nz;
      dst[9] = (_Float16)hvx; dst[10] = (_Float16)hvy; dst[11] = (_Float16)hvz;
      dst[12] = (_Float16)dvx; dst[13] = (_Float16)dvy; dst[14] = (_Float16)dvz;
#pragma unroll
      for (int j = 0; j < 64; ++j) dst[15 + j] = snaf[wv][j];
      dst[79] = (_Float16)ea;
#pragma unroll
      for (int j = 80; j < 96; ++j) dst[j] = (_Float16)0.0f;
    }
    __syncthreads();
    for (int ctl = 0; ctl < 8; ++ctl) {
      v8f c = {};
#pragma unroll
      for (int kc = 0; kc < 3; ++kc) {
        v16h aa = frag_a_lds(featbuf[wv], 96, kc * 32);
        v16h bb = frag_b_ldsT(sWb1T, 96, kc * 32, ctl * 16 + ncol);
        c = wmma16(aa, bb, c);
      }
      const int n = ctl * 16 + ncol;
      const float bias = bb1v[ctl];
#pragma unroll
      for (int r = 0; r < 8; ++r) {
        const float h = c[r] + bias;
        hbuf[wv][(hs * 8 + r) * 128 + n] = (_Float16)(h > 0.f ? h : 0.f);
      }
    }
    __syncthreads();
    {
      v8f c = {};
#pragma unroll
      for (int kc = 0; kc < 4; ++kc) {
        v16h aa = frag_a_lds(hbuf[wv], 128, kc * 32);
        v16h bb = frag_b_ldsT(sWb2T, 128, kc * 32, ncol);
        c = wmma16(aa, bb, c);
      }
      if (ncol < 3) {
#pragma unroll
        for (int r = 0; r < 8; ++r) {
          const int mr = t * 16 + hs * 8 + r;
          const float bwv = sigmoidf_(c[r] + bb2v);
          const float msk = (float)((bal >> mr) & 1ull);
          acc += bwv * (float)sIncoming[wv][mr][ncol] * msk;
        }
      }
    }
    __syncthreads();
  }

  // cross-half reduce: lane n (hs=0) + lane n+16 (hs=1) hold the two row halves
  const float other = __shfl_xor(acc, 16);
  if (lane < 3) {
    const float tinted = (acc + other) * invrc;
    out[idx * 3 + lane] += tinted;  // kernel 1 rewrote diffuse this launch -> deterministic
  }
}

// ========================= host-side launcher =========================
extern "C" void kernel_launch(void* const* d_in, const int* in_sizes, int n_in,
                              void* d_out, int out_size, void* d_ws, size_t ws_size,
                              hipStream_t stream) {
  const float* xyzs     = (const float*)d_in[0];
  const float* app      = (const float*)d_in[1];
  const float* viewdirs = (const float*)d_in[2];
  const float* normals  = (const float*)d_in[3];
  const float* noise    = (const float*)d_in[4];
  const float* u        = (const float*)d_in[5];
  const float* Wd1 = (const float*)d_in[6];  const float* bd1 = (const float*)d_in[7];
  const float* Wd2 = (const float*)d_in[8];  const float* bd2 = (const float*)d_in[9];
  const float* Wb1 = (const float*)d_in[10]; const float* bb1 = (const float*)d_in[11];
  const float* Wb2 = (const float*)d_in[12]; const float* bb2 = (const float*)d_in[13];
  const float* We1 = (const float*)d_in[14]; const float* be1 = (const float*)d_in[15];
  const float* We2 = (const float*)d_in[16]; const float* be2 = (const float*)d_in[17];
  const int* bounce_idx = (const int*)d_in[18];
  const unsigned char* ray_mask = (const unsigned char*)d_in[19];
  float* out   = (float*)d_out;
  float* r1_ws = (float*)d_ws;   // N floats of scratch for r1

  const int N = 131072, NB = 16384;

  diffuse_mlp_kernel<<<N / (K1_WAVES * 16), 128, 0, stream>>>(
      xyzs, viewdirs, app, Wd1, bd1, Wd2, bd2, out, r1_ws);

  bounce_kernel<<<NB / K2_WAVES, 128, 0, stream>>>(
      app, viewdirs, normals, noise, u, Wb1, bb1, Wb2, bb2,
      We1, be1, We2, be2, bounce_idx, ray_mask, r1_ws, out);
}
